// VQ_64931315581392
// MI455X (gfx1250) — matrix operand order repaired and checked by reading, not compiled, not run on hardware
//
#include <hip/hip_runtime.h>
#include <stdint.h>

// ---------- CDNA5 WMMA types ----------
typedef __attribute__((ext_vector_type(16))) __bf16 v16bf;
typedef __attribute__((ext_vector_type(8)))  float  v8f;

#define D 256
#define BETA 0.25f

__device__ __forceinline__ unsigned short f2bf(float f) {
  union { float f; unsigned u; } in; in.f = f;
  unsigned u = in.u;
  unsigned r = u + 0x7FFFu + ((u >> 16) & 1u);   // round-to-nearest-even
  return (unsigned short)(r >> 16);
}

// ---------------------------------------------------------------------------
// Kernel 1: codes f32 -> bf16 copy + squared norms.  One wave per code row.
// ---------------------------------------------------------------------------
__global__ void __launch_bounds__(256) vq_prep_codes(
    const float* __restrict__ codes, unsigned short* __restrict__ cbf,
    float* __restrict__ cnorm, int K) {
  int wave = (int)((blockIdx.x * blockDim.x + threadIdx.x) >> 5);
  int lane = threadIdx.x & 31;
  if (wave >= K) return;
  const float* crow = codes + (size_t)wave * D;
  unsigned short* brow = cbf + (size_t)wave * D;
  float s = 0.f;
#pragma unroll
  for (int d = lane; d < D; d += 32) {
    float c = crow[d];
    s += c * c;
    brow[d] = f2bf(c);
  }
#pragma unroll
  for (int off = 16; off >= 1; off >>= 1) s += __shfl_xor(s, off, 32);
  if (lane == 0) cnorm[wave] = s;
}

// ---------------------------------------------------------------------------
// Kernel 2: main VQ kernel.  256 threads = 8 wave32; each wave owns 16 rows
// of z.  Block double-buffers 16-code bf16 tiles (8KB) through LDS; each
// wave runs 8x v_wmma_f32_16x16x32_bf16 per tile (K=256) and tracks a
// register-resident (min, argmin) per output row.  Score = ||c||^2 - 2 z.c
// (||z||^2 dropped: constant per row, irrelevant for argmin).
// ---------------------------------------------------------------------------
__global__ void __launch_bounds__(256) vq_main(
    const float* __restrict__ z, const float* __restrict__ codes,
    const unsigned short* __restrict__ cbf, const float* __restrict__ cnorm,
    float* __restrict__ out_z, float* __restrict__ out_loss,
    float* __restrict__ out_idx, int K) {
  __shared__ unsigned short btile[2][16 * D];      // 2 x 8KB double buffer

  const int tid    = threadIdx.x;
  const int lane   = tid & 31;
  const int half   = lane >> 4;                    // 0: lanes 0-15, 1: 16-31
  const int nloc   = lane & 15;
  const int rowBase = (blockIdx.x * 8 + (tid >> 5)) * 16;

  // ---- build A fragments: 16 z rows x K=256 as bf16, held in registers ----
  // A-layout used here (contiguous k, same as B): lane<16 holds row=lane, K = c*32+{0..15};
  // lane>=16 holds row=lane-16, K = c*32+{16..31}.
  const float* zrow = z + (size_t)(rowBase + nloc) * D;
  v16bf za[8];
#pragma unroll
  for (int c = 0; c < 8; ++c) {
    union { v16bf v; unsigned short s[16]; } a;
    int k0 = c * 32 + half * 16;
#pragma unroll
    for (int j = 0; j < 8; ++j) a.s[j]     = f2bf(zrow[k0 + j]);
#pragma unroll
    for (int j = 0; j < 8; ++j) a.s[8 + j] = f2bf(zrow[k0 + 8 + j]);
    za[c] = a.v;
  }

  float best[8];
  int   bidx[8];
#pragma unroll
  for (int r = 0; r < 8; ++r) { best[r] = 3.4e38f; bidx[r] = 0; }

  const int ntiles = K >> 4;                       // 64 tiles of 16 codes
  const uint4* src = (const uint4*)cbf;            // tile = 512 uint4 (8KB)

  // prefetch tile 0 into buffer 0 (each thread stages 32B)
  uint4 p0 = src[tid * 2];
  uint4 p1 = src[tid * 2 + 1];
  ((uint4*)btile[0])[tid * 2]     = p0;
  ((uint4*)btile[0])[tid * 2 + 1] = p1;
  __syncthreads();

  for (int t = 0; t < ntiles; ++t) {
    const int cur = t & 1;
    if (t + 1 < ntiles) {                          // global prefetch of next tile
      p0 = src[(t + 1) * 512 + tid * 2];
      p1 = src[(t + 1) * 512 + tid * 2 + 1];
    }

    // ---- B fragment from LDS: lane = column (code) nloc.  B-layout used here (contiguous k, as A):
    // lane<16 holds K=0..15 (32 contiguous bytes), lane>=16 holds K=16..31.
    const unsigned short* brow = &btile[cur][nloc * D];
    const float cn = cnorm[t * 16 + nloc];

    v8f acc = {0.f, 0.f, 0.f, 0.f, 0.f, 0.f, 0.f, 0.f};
#pragma unroll
    for (int c = 0; c < 8; ++c) {
      union { v16bf v; uint4 q[2]; } b;
      const uint4* bp = (const uint4*)(brow + c * 32 + half * 16);
      b.q[0] = bp[0];
      b.q[1] = bp[1];
      acc = __builtin_amdgcn_wmma_f32_16x16x32_bf16(
          /*neg_a=*/false, za[c], /*neg_b=*/false, b.v,
          /*c_mod=*/(short)0, acc, /*reuse_a=*/false, /*reuse_b=*/false);
    }

    // C/D layout: acc[r] holds (M=r, N=nloc) for lanes 0-15, (M=r+8, N=nloc)
    // for lanes 16-31.  Update per-row running argmin.
    const int nglob = t * 16 + nloc;
#pragma unroll
    for (int r = 0; r < 8; ++r) {
      float dist = cn - 2.0f * acc[r];
      if (dist < best[r]) { best[r] = dist; bidx[r] = nglob; }
    }

    __syncthreads();                               // everyone done with buffers
    if (t + 1 < ntiles) {
      ((uint4*)btile[cur ^ 1])[tid * 2]     = p0;
      ((uint4*)btile[cur ^ 1])[tid * 2 + 1] = p1;
    }
    __syncthreads();                               // next tile visible
  }

  // ---- argmin butterfly inside each 16-lane half (xor 1..8 stays in half) --
#pragma unroll
  for (int r = 0; r < 8; ++r) {
    float v = best[r]; int i = bidx[r];
#pragma unroll
    for (int off = 1; off < 16; off <<= 1) {
      float ov = __shfl_xor(v, off, 32);
      int   oi = __shfl_xor(i, off, 32);
      if (ov < v || (ov == v && oi < i)) { v = ov; i = oi; }  // ties -> low idx
    }
    best[r] = v; bidx[r] = i;
  }

  // ---- output pass: gather winning code row, straight-through add, loss ----
  for (int m = 0; m < 16; ++m) {
    const int idx  = __shfl(bidx[m & 7], (m >> 3) << 4, 32);
    const int grow = rowBase + m;
    const float* crow = codes + (size_t)idx * D;   // f32 codes (full precision)
    const float* zr   = z + (size_t)grow * D;
    float* orow = out_z + (size_t)grow * D;
    float ssum = 0.f;
#pragma unroll
    for (int d = lane; d < D; d += 32) {
      float c    = crow[d];
      float zz   = zr[d];
      float diff = c - zz;                         // (z_q - z)
      ssum += diff * diff;
      orow[d] = zz + diff;                         // z + sg(z_q - z), bit-match
    }
#pragma unroll
    for (int off = 1; off < 32; off <<= 1) ssum += __shfl_xor(ssum, off, 32);
    if (lane == 0) {
      out_loss[grow] = (1.0f + BETA) * ssum * (1.0f / (float)D);
      out_idx[grow]  = (float)idx;
    }
  }
}

// ---------------------------------------------------------------------------
extern "C" void kernel_launch(void* const* d_in, const int* in_sizes, int n_in,
                              void* d_out, int out_size, void* d_ws,
                              size_t ws_size, hipStream_t stream) {
  const float* z     = (const float*)d_in[0];
  const float* codes = (const float*)d_in[1];
  const int N = in_sizes[0] / D;                   // 131072
  const int K = in_sizes[1] / D;                   // 1024

  // workspace: [cnorm f32 K][pad][codes bf16 K*D]
  float* cnorm = (float*)d_ws;
  unsigned short* cbf =
      (unsigned short*)((char*)d_ws + (((size_t)K * sizeof(float) + 255) & ~(size_t)255));

  float* out_z    = (float*)d_out;
  float* out_loss = out_z + (size_t)N * D;
  float* out_idx  = out_loss + N;

  vq_prep_codes<<<K / 8, 256, 0, stream>>>(codes, cbf, cnorm, K);
  vq_main<<<N / 128, 256, 0, stream>>>(z, codes, cbf, cnorm,
                                       out_z, out_loss, out_idx, K);
}
